// V2EMsgPass_40303973106022
// MI455X (gfx1250) — matrix operand order repaired and checked by reading, not compiled, not run on hardware
//
#include <hip/hip_runtime.h>

typedef _Float16 v2h  __attribute__((ext_vector_type(2)));
typedef __fp16   v2hp __attribute__((ext_vector_type(2)));   // return type of cvt_pkrtz
typedef _Float16 v4h  __attribute__((ext_vector_type(4)));
typedef _Float16 v8h  __attribute__((ext_vector_type(8)));
typedef _Float16 v16h __attribute__((ext_vector_type(16)));
typedef float    v8f  __attribute__((ext_vector_type(8)));

#define TILE_E 32
#define BK     32
#define DDIM   128
#define LROW   36   // padded LDS row (halves): 72 B -> 8B-aligned b64 reads, bank-spread stores

__device__ __forceinline__ v16h cat4(v4h a, v4h b, v4h c, v4h d) {
    v8h lo = __builtin_shufflevector(a, b, 0, 1, 2, 3, 4, 5, 6, 7);
    v8h hi = __builtin_shufflevector(c, d, 0, 1, 2, 3, 4, 5, 6, 7);
    return __builtin_shufflevector(lo, hi,
            0, 1, 2, 3, 4, 5, 6, 7, 8, 9, 10, 11, 12, 13, 14, 15);
}

__global__ __launch_bounds__(256)
void v2e_msgpass_kernel(const float* __restrict__ X,
                        const float* __restrict__ H,
                        float* __restrict__ out,
                        int N, int E)
{
    __shared__ __align__(16) _Float16 lA[TILE_E][LROW];  // [e][n] f16, padded
    __shared__ __align__(16) _Float16 lX[DDIM][LROW];    // [d][n] f16, padded
    __shared__ float sdeg[TILE_E];

    const int tid  = threadIdx.x;
    const int lane = tid & 31;
    const int wave = tid >> 5;        // 0..7
    const int we   = wave >> 2;       // 0..1 : e-subtile
    const int wd   = wave & 3;        // 0..3 : d-subtile
    const int e0   = blockIdx.x * TILE_E;
    const int d0   = wd * 32;

    const int m    = lane & 15;
    const int hgrp = lane >> 4;

    if (tid < TILE_E) sdeg[tid] = 0.0f;

    // H staging: 32n x 32e tile; thread owns n-pair hp and e-pair he (coalesced float2 rows)
    const int hp = tid >> 4;          // 0..15 (n-pair)
    const int he = (tid & 15) * 2;    // 0..30 (e base)
    // X staging: 32n x 128d tile; thread owns 2 n-pairs and 4 d columns (coalesced float4 rows)
    const int xp = tid >> 5;          // 0..7  (n-pair base)
    const int xd = (tid & 31) * 4;    // 0..124

    float deg0 = 0.f, deg1 = 0.f;
    v8f acc0 = {};
    v8f acc1 = {};

    for (int k = 0; k < N; k += BK) {
        // ---- stage H (HBM stream) as f16, [e][n], + degree partials ----
        const float* hrow = H + (size_t)(k + 2 * hp) * E + e0 + he;
        float2 ha = *(const float2*)hrow;         // n = 2hp
        float2 hb = *(const float2*)(hrow + E);   // n = 2hp+1
        deg0 += ha.x + hb.x;
        deg1 += ha.y + hb.y;
        *(v2hp*)&lA[he    ][2 * hp] = __builtin_amdgcn_cvt_pkrtz(ha.x, hb.x);
        *(v2hp*)&lA[he + 1][2 * hp] = __builtin_amdgcn_cvt_pkrtz(ha.y, hb.y);

        // ---- stage X (L2 resident) as f16, [d][n] ----
        #pragma unroll
        for (int r = 0; r < 2; ++r) {
            int p = xp + r * 8;                               // n-pair 0..15
            const float* xrow = X + (size_t)(k + 2 * p) * DDIM + xd;
            float4 xa = *(const float4*)xrow;                 // n = 2p
            float4 xb = *(const float4*)(xrow + DDIM);        // n = 2p+1
            *(v2hp*)&lX[xd + 0][2 * p] = __builtin_amdgcn_cvt_pkrtz(xa.x, xb.x);
            *(v2hp*)&lX[xd + 1][2 * p] = __builtin_amdgcn_cvt_pkrtz(xa.y, xb.y);
            *(v2hp*)&lX[xd + 2][2 * p] = __builtin_amdgcn_cvt_pkrtz(xa.z, xb.z);
            *(v2hp*)&lX[xd + 3][2 * p] = __builtin_amdgcn_cvt_pkrtz(xa.w, xb.w);
        }

        if (k + BK < N)
            __builtin_prefetch(H + (size_t)(k + BK + 2 * hp) * E + e0 + he, 0, 1);

        __syncthreads();

        // ---- A fragment (16x32 f16): row we*16+m, K = kb..kb+15 (contiguous) ----
        const int kb = hgrp * 16;
        const _Float16* ar = &lA[we * 16 + m][0];
        v16h a = cat4(*(const v4h*)(ar + kb),
                      *(const v4h*)(ar + kb + 4),
                      *(const v4h*)(ar + kb + 8),
                      *(const v4h*)(ar + kb + 12));

        // ---- B fragments (32x16 f16): col d, K = hgrp*16 .. hgrp*16+15 (contiguous) ----
        const int kB = hgrp * 16;
        const _Float16* b0r = &lX[d0 + m][kB];
        v16h b0 = cat4(*(const v4h*)(b0r),
                       *(const v4h*)(b0r + 4),
                       *(const v4h*)(b0r + 8),
                       *(const v4h*)(b0r + 12));
        const _Float16* b1r = &lX[d0 + 16 + m][kB];
        v16h b1 = cat4(*(const v4h*)(b1r),
                       *(const v4h*)(b1r + 4),
                       *(const v4h*)(b1r + 8),
                       *(const v4h*)(b1r + 12));

        acc0 = __builtin_amdgcn_wmma_f32_16x16x32_f16(
                   false, a, false, b0, (short)0, acc0, false, false);
        acc1 = __builtin_amdgcn_wmma_f32_16x16x32_f16(
                   false, a, false, b1, (short)0, acc1, false, false);

        __syncthreads();
    }

    // ---- degree reduction (each H element counted exactly once across threads) ----
    atomicAdd(&sdeg[he],     deg0);
    atomicAdd(&sdeg[he + 1], deg1);
    __syncthreads();

    // ---- scale by 1/deg (0-degree -> 0) and store ----
    // C layout: VGPR v -> M = hgrp*8 + v (within 16-row tile), lane column = m
    #pragma unroll
    for (int v = 0; v < 8; ++v) {
        int   mr  = we * 16 + hgrp * 8 + v;
        float dg  = sdeg[mr];
        float inv = (dg != 0.0f) ? (1.0f / dg) : 0.0f;
        out[(size_t)(e0 + mr) * DDIM + d0 + m]      = acc0[v] * inv;
        out[(size_t)(e0 + mr) * DDIM + d0 + 16 + m] = acc1[v] * inv;
    }
}

extern "C" void kernel_launch(void* const* d_in, const int* in_sizes, int n_in,
                              void* d_out, int out_size, void* d_ws, size_t ws_size,
                              hipStream_t stream) {
    (void)n_in; (void)d_ws; (void)ws_size; (void)out_size;
    const float* X = (const float*)d_in[0];   // [N, 128]
    const float* H = (const float*)d_in[1];   // [N, E]
    float* out = (float*)d_out;               // [E, 128]

    const int N = in_sizes[0] / DDIM;         // 8192
    const int E = in_sizes[1] / N;            // 8192

    dim3 grid(E / TILE_E);                    // 256 blocks (32-row e-tiles)
    dim3 block(256);                          // 8 wave32
    v2e_msgpass_kernel<<<grid, block, 0, stream>>>(X, H, out, N, E);
}
